// QINCoStep_386547057401
// MI455X (gfx1250) — compile-verified, matrix operand order checked by reading, not run on hardware
//
#include <hip/hip_runtime.h>
#include <hip/hip_bf16.h>

// ---------------------------------------------------------------------------
// QINCo step, fused for MI455X (gfx1250, wave32, WMMA + TDM).
// Shapes: D=128, K=256, HID=256, NBLK=2, N=4096.
// One workgroup per n; C (K x D) in LDS fp32 (stride 132), h in LDS bf16;
// all 4 GEMMs per n via v_wmma_f32_16x16x32_bf16 (fp32 accum).
// C-init tile is DMA'd by the Tensor Data Mover (tensor_load_to_lds) with
// hardware padding producing the conflict-free LDS stride directly.
// ~275 GFLOP total vs ~15 MB HBM traffic -> compute bound on the WMMA pipe.
// ---------------------------------------------------------------------------

typedef __attribute__((ext_vector_type(16))) __bf16 v16bf;
typedef __attribute__((ext_vector_type(8)))  float  v8f;
typedef unsigned int uint32x4 __attribute__((ext_vector_type(4)));
typedef int          int32x8  __attribute__((ext_vector_type(8)));
typedef int          int32x4  __attribute__((ext_vector_type(4)));

#define Dm    128
#define Kc    256
#define HIDm  256
#define NBLKm 2
#define Nm    4096

#define CSTR 132   // fp32 row stride for C in LDS (128 data + 4 pad DWORDs)
#define HSTR 264   // bf16 row stride for h in LDS

// ---------------------------- prep kernels ---------------------------------

__global__ void prep_weights_bf16(const float* __restrict__ W1,
                                  const float* __restrict__ W2,
                                  __bf16* __restrict__ W1bf,
                                  __bf16* __restrict__ W2bf, int n) {
    int i = blockIdx.x * blockDim.x + threadIdx.x;
    if (i < n) {
        W1bf[i] = (__bf16)W1[i];
        W2bf[i] = (__bf16)W2[i];
    }
}

// c_part[k][d] = sum_j cb[k][j] * Wp[d][D + j] + bp[d]
__global__ void prep_cpart(const float* __restrict__ cb,
                           const float* __restrict__ Wp,
                           const float* __restrict__ bp,
                           float* __restrict__ c_part) {
    int k = blockIdx.x, d = threadIdx.x;
    __shared__ float row[Dm];
    row[d] = cb[k * Dm + d];
    __syncthreads();
    float acc = bp[d];
    const float* wrow = Wp + d * (2 * Dm) + Dm;
    #pragma unroll 8
    for (int j = 0; j < Dm; ++j) acc += row[j] * wrow[j];
    c_part[k * Dm + d] = acc;
}

// x_part[n][d] = sum_j x_hat[n][j] * Wp[d][j]
__global__ void prep_xpart(const float* __restrict__ x_hat,
                           const float* __restrict__ Wp,
                           float* __restrict__ x_part) {
    int n = blockIdx.x, d = threadIdx.x;
    __shared__ float row[Dm];
    row[d] = x_hat[n * Dm + d];
    __syncthreads();
    float acc = 0.f;
    const float* wrow = Wp + d * (2 * Dm);
    #pragma unroll 8
    for (int j = 0; j < Dm; ++j) acc += row[j] * wrow[j];
    x_part[n * Dm + d] = acc;
}

// ----------------------------- main kernel ---------------------------------

__global__ void __launch_bounds__(256, 1)
qinco_main(const float* __restrict__ x, const float* __restrict__ x_hat,
           const float* __restrict__ x_part, const float* __restrict__ c_part,
           const __bf16* __restrict__ W1bf, const float* __restrict__ b1,
           const __bf16* __restrict__ W2bf, const float* __restrict__ b2,
           float* __restrict__ out_codes, float* __restrict__ out_xhat,
           float* __restrict__ out_r,     float* __restrict__ out_csel) {
    extern __shared__ char smem[];
    float*  Cs = (float*)smem;                                        // [Kc][CSTR]
    __bf16* Hs = (__bf16*)(smem + (size_t)Kc * CSTR * sizeof(float)); // [Kc][HSTR]

    const int n    = blockIdx.x;
    const int tid  = threadIdx.x;
    const int lane = tid & 31;
    const int wave = tid >> 5;        // 0..7, owns M rows [32*wave, 32*wave+32)
    const int lrow = lane & 15;       // WMMA 16-bit layout: row/col = lane&15
    const int hi   = lane >> 4;       // lanes 16..31 get K-base +8 / M-base +8
    const int mt0  = wave * 2;        // first 16-row M tile of this wave

    const float* xp = x_part + n * Dm;

    // ---- C = c_part (TDM DMA into padded LDS) + x_part[n] ------------------
#if defined(__gfx1250__) && __has_builtin(__builtin_amdgcn_tensor_load_to_lds)
    if (wave == 0) {
        unsigned long long ga = (unsigned long long)c_part;
        uint32x4 g0;
        g0[0] = 1u;                                   // count=1, user mode
        g0[1] = 0u;                                   // lds_addr = 0 (Cs base)
        g0[2] = (unsigned int)(ga & 0xFFFFFFFFu);     // global_addr[31:0]
        g0[3] = (unsigned int)((ga >> 32) & 0x01FFFFFFu) | (2u << 30); // +type=2
        int32x8 g1;
        g1[0] = (2 << 16)       // data_size = 4 bytes
              | (1 << 20)       // pad_enable
              | (6 << 22)       // pad_interval: every 128 DWORDs
              | (3 << 25);      // pad_amount: 4 DWORDs  -> LDS stride 132
        g1[1] = (Dm & 0xFFFF) << 16;   // tensor_dim0 = 128  (bits 63:48)
        g1[2] = (Kc & 0xFFFF) << 16;   // tensor_dim1 = 256  (bits 95:80)
        g1[3] = (Dm & 0xFFFF) << 16;   // tile_dim0  = 128  (bits 127:112)
        g1[4] = (Kc & 0xFFFF);         // tile_dim1  = 256  (bits 143:128)
        g1[5] = Dm;                    // tensor_dim0_stride = 128 (dense)
        g1[6] = 0;
        g1[7] = 0;
        int32x4 gz = {0, 0, 0, 0};
#if __clang_major__ >= 23
        int32x8 gz8 = {0, 0, 0, 0, 0, 0, 0, 0};
        __builtin_amdgcn_tensor_load_to_lds(g0, g1, gz, gz, gz8, 0);
#else
        __builtin_amdgcn_tensor_load_to_lds(g0, g1, gz, gz, 0);
#endif
#if __has_builtin(__builtin_amdgcn_s_wait_tensorcnt)
        __builtin_amdgcn_s_wait_tensorcnt(0);
#endif
    }
    __syncthreads();
    for (int idx = tid; idx < Kc * Dm; idx += 256) {
        int k = idx >> 7, d = idx & 127;
        Cs[k * CSTR + d] += xp[d];
    }
#else
    for (int idx = tid; idx < Kc * Dm; idx += 256) {
        int k = idx >> 7, d = idx & 127;
        Cs[k * CSTR + d] = xp[d] + c_part[idx];
    }
#endif
    __syncthreads();

    // ---- residual MLP blocks ----------------------------------------------
    for (int blk = 0; blk < NBLKm; ++blk) {
        const __bf16* W1  = W1bf + blk * HIDm * Dm;   // (HID, D) row-major
        const __bf16* W2  = W2bf + blk * Dm * HIDm;   // (D, HID) row-major
        const float*  bb1 = b1 + blk * HIDm;
        const float*  bb2 = b2 + blk * Dm;

        // warm caches for the second GEMM's weights while doing GEMM1
        __builtin_prefetch(W2 + ((tid * 64) & (Dm * HIDm - 1)), 0, 1);

        // GEMM1: Hs = relu(Cs(256x128) @ W1^T(128x256) + b1)
        for (int nt = 0; nt < 16; ++nt) {
            const float bias = bb1[nt * 16 + lrow];
            v8f acc0, acc1;
            #pragma unroll
            for (int v = 0; v < 8; ++v) { acc0[v] = bias; acc1[v] = bias; }
            #pragma unroll
            for (int kt = 0; kt < 4; ++kt) {
                const int kb = kt * 32 + hi * 8;
                v16bf a0, a1, b;
                const float*  pa0 = Cs + (mt0 * 16 + lrow) * CSTR + kb;
                const float*  pa1 = Cs + ((mt0 + 1) * 16 + lrow) * CSTR + kb;
                const __bf16* pb  = W1 + (nt * 16 + lrow) * Dm + kb;
                #pragma unroll
                for (int e = 0; e < 8; ++e) {
                    a0[e]     = (__bf16)pa0[e];
                    a0[8 + e] = (__bf16)pa0[16 + e];
                    a1[e]     = (__bf16)pa1[e];
                    a1[8 + e] = (__bf16)pa1[16 + e];
                    b[e]      = pb[e];
                    b[8 + e]  = pb[16 + e];
                }
                acc0 = __builtin_amdgcn_wmma_f32_16x16x32_bf16(
                           false, a0, false, b, (short)0, acc0, false, false);
                acc1 = __builtin_amdgcn_wmma_f32_16x16x32_bf16(
                           false, a1, false, b, (short)0, acc1, false, false);
            }
            const int col = nt * 16 + lrow;
            #pragma unroll
            for (int v = 0; v < 8; ++v) {
                float t0 = acc0[v] > 0.f ? acc0[v] : 0.f;
                float t1 = acc1[v] > 0.f ? acc1[v] : 0.f;
                Hs[(mt0 * 16 + hi * 8 + v) * HSTR + col]       = (__bf16)t0;
                Hs[((mt0 + 1) * 16 + hi * 8 + v) * HSTR + col] = (__bf16)t1;
            }
        }
        __syncthreads();

        // GEMM2: Cs += Hs(256x256) @ W2^T(256x128) + b2
        for (int nt = 0; nt < 8; ++nt) {
            const float bias = bb2[nt * 16 + lrow];
            v8f acc0, acc1;
            #pragma unroll
            for (int v = 0; v < 8; ++v) { acc0[v] = bias; acc1[v] = bias; }
            #pragma unroll
            for (int kt = 0; kt < 8; ++kt) {
                const int kb = kt * 32 + hi * 8;
                v16bf a0, a1, b;
                const __bf16* pa0 = Hs + (mt0 * 16 + lrow) * HSTR + kb;
                const __bf16* pa1 = Hs + ((mt0 + 1) * 16 + lrow) * HSTR + kb;
                const __bf16* pb  = W2 + (nt * 16 + lrow) * HIDm + kb;
                #pragma unroll
                for (int e = 0; e < 8; ++e) {
                    a0[e]     = pa0[e];
                    a0[8 + e] = pa0[16 + e];
                    a1[e]     = pa1[e];
                    a1[8 + e] = pa1[16 + e];
                    b[e]      = pb[e];
                    b[8 + e]  = pb[16 + e];
                }
                acc0 = __builtin_amdgcn_wmma_f32_16x16x32_bf16(
                           false, a0, false, b, (short)0, acc0, false, false);
                acc1 = __builtin_amdgcn_wmma_f32_16x16x32_bf16(
                           false, a1, false, b, (short)0, acc1, false, false);
            }
            const int col = nt * 16 + lrow;
            #pragma unroll
            for (int v = 0; v < 8; ++v) {
                Cs[(mt0 * 16 + hi * 8 + v) * CSTR + col]       += acc0[v];
                Cs[((mt0 + 1) * 16 + hi * 8 + v) * CSTR + col] += acc1[v];
            }
        }
        __syncthreads();
    }

    // ---- distances, argmin, outputs ---------------------------------------
    float* rl   = (float*)Hs;        // overlay on Hs (no longer needed)
    float* dist = rl + Dm;           // 256 floats
    int*   didx = (int*)(dist + Kc); // 256 ints

    if (tid < Dm) {
        float rv = x[n * Dm + tid] - x_hat[n * Dm + tid];
        rl[tid] = rv;
        out_r[n * Dm + tid] = rv;
    }
    __syncthreads();

    {
        float ds = 0.f;
        const float* Crow = Cs + tid * CSTR;
        #pragma unroll 8
        for (int d = 0; d < Dm; ++d) {
            float df = rl[d] - Crow[d];
            ds += df * df;
        }
        dist[tid] = ds;
        didx[tid] = tid;
    }
    __syncthreads();

    for (int s = 128; s > 0; s >>= 1) {
        if (tid < s) {
            if (dist[tid + s] < dist[tid]) {
                dist[tid] = dist[tid + s];
                didx[tid] = didx[tid + s];
            }
        }
        __syncthreads();
    }

    const int kmin = didx[0];
    if (tid == 0) out_codes[n] = (float)kmin;
    if (tid < Dm) {
        float cv = Cs[kmin * CSTR + tid];
        out_csel[n * Dm + tid] = cv;
        out_xhat[n * Dm + tid] = x_hat[n * Dm + tid] + cv;
    }
}

// ------------------------------ launcher -----------------------------------

extern "C" void kernel_launch(void* const* d_in, const int* in_sizes, int n_in,
                              void* d_out, int out_size, void* d_ws, size_t ws_size,
                              hipStream_t stream) {
    const float* x     = (const float*)d_in[0];
    const float* x_hat = (const float*)d_in[1];
    const float* cb    = (const float*)d_in[2];
    const float* Wp    = (const float*)d_in[3];
    const float* bp    = (const float*)d_in[4];
    const float* W1    = (const float*)d_in[5];
    const float* b1    = (const float*)d_in[6];
    const float* W2    = (const float*)d_in[7];
    const float* b2    = (const float*)d_in[8];

    char* ws = (char*)d_ws;
    float*  x_part = (float*)ws;  ws += (size_t)Nm * Dm * sizeof(float);
    float*  c_part = (float*)ws;  ws += (size_t)Kc * Dm * sizeof(float);
    __bf16* W1bf   = (__bf16*)ws; ws += (size_t)NBLKm * HIDm * Dm * sizeof(__bf16);
    __bf16* W2bf   = (__bf16*)ws; ws += (size_t)NBLKm * Dm * HIDm * sizeof(__bf16);

    float* out       = (float*)d_out;
    float* out_codes = out;                  // N
    float* out_xhat  = out_codes + Nm;       // N*D
    float* out_r     = out_xhat + Nm * Dm;   // N*D
    float* out_csel  = out_r + Nm * Dm;      // N*D

    const int nw = NBLKm * HIDm * Dm;
    prep_weights_bf16<<<(nw + 255) / 256, 256, 0, stream>>>(W1, W2, W1bf, W2bf, nw);
    prep_cpart<<<Kc, Dm, 0, stream>>>(cb, Wp, bp, c_part);
    prep_xpart<<<Nm, Dm, 0, stream>>>(x_hat, Wp, x_part);

    size_t shmem = (size_t)Kc * CSTR * sizeof(float) + (size_t)Kc * HSTR * sizeof(__bf16);
    qinco_main<<<Nm, 256, shmem, stream>>>(x, x_hat, x_part, c_part,
                                           W1bf, b1, W2bf, b2,
                                           out_codes, out_xhat, out_r, out_csel);
}
